// PointNet2FeatureExtractorWithFP_6700148982226
// MI455X (gfx1250) — matrix operand order repaired and checked by reading, not compiled, not run on hardware
//
#include <hip/hip_runtime.h>
#include <hip/hip_bf16.h>
#include <stdint.h>

// ---------------------------------------------------------------------------
// PointNet++ (SSG) pipeline for MI455X (gfx1250, wave32, WMMA).
// Activations are kept as bf16, K-padded to multiples of 32, so the WMMA GEMM
// stages tiles with branch-free async global->LDS copies (ASYNCcnt path) and
// runs 4 x v_wmma_f32_16x16x32_bf16 per K-tile per wave, double-buffered.
// ---------------------------------------------------------------------------

typedef __attribute__((ext_vector_type(16))) __bf16 v16bf_t;
typedef __attribute__((ext_vector_type(8)))  float  v8f_t;

__device__ __forceinline__ unsigned short f2bf(float f) {
    unsigned u = __float_as_uint(f);
    u += 0x7FFFu + ((u >> 16) & 1u);       // round-to-nearest-even
    return (unsigned short)(u >> 16);
}
__device__ __forceinline__ float bf2f(unsigned short u) {
    return __uint_as_float(((unsigned)u) << 16);
}

// Async global -> LDS copy, 8 bytes per lane (gfx1250, tracked by ASYNCcnt).
__device__ __forceinline__ void async_copy_b64(unsigned lds_off, const void* gaddr) {
    asm volatile("global_load_async_to_lds_b64 %0, %1, off"
                 :: "v"(lds_off), "v"(gaddr) : "memory");
}

// ---------------------------------------------------------------------------
// Farthest point sampling: one block per batch, dist kept in registers.
// ---------------------------------------------------------------------------
template <int NPTS>
__global__ void fps_kernel(const float* __restrict__ xyz, int* __restrict__ out, int S) {
    constexpr int PERT = NPTS / 1024;
    const int b = blockIdx.x;
    const float* px = xyz + (size_t)b * NPTS * 3;
    float dist[PERT];
#pragma unroll
    for (int j = 0; j < PERT; ++j) dist[j] = 1e10f;

    __shared__ float sval[1024];
    __shared__ int   sidx[1024];
    __shared__ float cpt[3];

    int cur = 0;
    for (int i = 0; i < S; ++i) {
        if (threadIdx.x == 0) out[b * S + i] = cur;
        if (threadIdx.x < 3) cpt[threadIdx.x] = px[(size_t)cur * 3 + threadIdx.x];
        __syncthreads();
        float cx = cpt[0], cy = cpt[1], cz = cpt[2];
        float bv = -1.0f; int bi = 0;
#pragma unroll
        for (int j = 0; j < PERT; ++j) {
            int t = threadIdx.x + j * 1024;
            float dx = px[t * 3 + 0] - cx;
            float dy = px[t * 3 + 1] - cy;
            float dz = px[t * 3 + 2] - cz;
            float d  = dx * dx + dy * dy + dz * dz;
            float nd = fminf(dist[j], d);
            dist[j] = nd;
            if (nd > bv) { bv = nd; bi = t; }
        }
        sval[threadIdx.x] = bv; sidx[threadIdx.x] = bi;
        __syncthreads();
        for (int off = 512; off > 0; off >>= 1) {
            if ((int)threadIdx.x < off) {
                float ov = sval[threadIdx.x + off];
                int   oi = sidx[threadIdx.x + off];
                if (ov > sval[threadIdx.x] ||
                    (ov == sval[threadIdx.x] && oi < sidx[threadIdx.x])) {
                    sval[threadIdx.x] = ov; sidx[threadIdx.x] = oi;
                }
            }
            __syncthreads();
        }
        cur = sidx[0];
        __syncthreads();
    }
}

__global__ void gather_xyz_kernel(const float* __restrict__ xyz, const int* __restrict__ fidx,
                                  float* __restrict__ out, int Npts, int S) {
    int p = blockIdx.x * blockDim.x + threadIdx.x;
    if (p >= 2 * S) return;
    int b = p / S;
    int i = fidx[p];
    const float* src = xyz + ((size_t)b * Npts + i) * 3;
    out[(size_t)p * 3 + 0] = src[0];
    out[(size_t)p * 3 + 1] = src[1];
    out[(size_t)p * 3 + 2] = src[2];
}

__global__ void ball_query_kernel(const float* __restrict__ xyz, const float* __restrict__ nxyz,
                                  int* __restrict__ gidx, int Npts, int S, int K, float r2) {
    int p = blockIdx.x * blockDim.x + threadIdx.x;
    if (p >= 2 * S) return;
    int b = p / S;
    const float* pc = nxyz + (size_t)p * 3;
    float cx = pc[0], cy = pc[1], cz = pc[2];
    const float* px = xyz + (size_t)b * Npts * 3;
    int* out = gidx + (size_t)p * K;
    int cnt = 0, first = 0;
    for (int t = 0; t < Npts && cnt < K; ++t) {
        float dx = px[t * 3 + 0] - cx;
        float dy = px[t * 3 + 1] - cy;
        float dz = px[t * 3 + 2] - cz;
        if (dx * dx + dy * dy + dz * dz <= r2) {
            if (cnt == 0) first = t;
            out[cnt++] = t;
        }
    }
    for (int k = cnt; k < K; ++k) out[k] = first;
}

// Build grouped GEMM rows (bf16, K-padded): [centered xyz | feats | 0-pad].
__global__ void gather_group_kernel(const float* __restrict__ xyz, const float* __restrict__ nxyz,
                                    const int* __restrict__ gidx,
                                    const unsigned short* __restrict__ feats,
                                    unsigned short* __restrict__ Xb,
                                    int Npts, int S, int K, int Cf, int Kp) {
    size_t rows = (size_t)2 * S * K;
    size_t row = (size_t)blockIdx.x * blockDim.x + threadIdx.x;
    if (row >= rows) return;
    size_t bs = row / K;
    int b = (int)(bs / S);
    int g = gidx[row];
    const float* pg = xyz + ((size_t)b * Npts + g) * 3;
    const float* pc = nxyz + bs * 3;
    unsigned short* xo = Xb + row * Kp;
    xo[0] = f2bf(pg[0] - pc[0]);
    xo[1] = f2bf(pg[1] - pc[1]);
    xo[2] = f2bf(pg[2] - pc[2]);
    const unsigned short* pf = feats + ((size_t)b * Npts + g) * Cf;
    for (int c = 0; c < Cf; ++c) xo[3 + c] = pf[c];
    for (int c = 3 + Cf; c < Kp; ++c) xo[c] = 0;
}

// Pack fp32 weights [Nc,Kc] -> bf16 [Nc,Kp] zero-padded.
__global__ void pack_weight_kernel(const float* __restrict__ W, unsigned short* __restrict__ Wb,
                                   int Nc, int Kc, int Kp) {
    size_t i = (size_t)blockIdx.x * blockDim.x + threadIdx.x;
    if (i >= (size_t)Nc * Kp) return;
    int n = (int)(i / Kp), k = (int)(i % Kp);
    Wb[i] = (k < Kc) ? f2bf(W[(size_t)n * Kc + k]) : (unsigned short)0;
}

// ---------------------------------------------------------------------------
// WMMA GEMM, fused bias+ReLU:  Y[M,Nc] = relu(X[M,Kp] @ W[Nc,Kp]^T + b)
// Block = 256 threads (8 wave32s), macro-tile 128(M) x 64(N), K-tile 32.
// Each wave: 32x32 output = 4 accumulators, 2 A-frags x 2 B-frags.
// LDS double-buffered, filled with async b64 copies (6 per wave per tile).
// Requires: M % 128 == 0, Nc % 64 == 0, Kp % 32 == 0 (guaranteed by caller).
// ---------------------------------------------------------------------------
#define GEMM_BM 128
#define GEMM_BN 64
#define GEMM_KT 32
#define GEMM_XS 36   // shorts per LDS row (72B = 18 banks -> conflict-free)

__global__ void __launch_bounds__(256)
gemm_bias_relu_wmma(const unsigned short* __restrict__ Xb, const unsigned short* __restrict__ Wb,
                    const float* __restrict__ bias, unsigned short* __restrict__ Yb,
                    int M, int Nc, int Kp) {
    __shared__ unsigned short lx[2][GEMM_BM * GEMM_XS];
    __shared__ unsigned short lw[2][GEMM_BN * GEMM_XS];

    const int tid  = threadIdx.x;
    const int wave = tid >> 5;
    const int lane = tid & 31;
    const int g    = lane >> 4;
    const int hl   = lane & 15;
    const int wm   = (wave & 3) * 32;      // wave M offset inside panel
    const int wn   = (wave >> 2) * 32;     // wave N offset inside panel
    const int m0   = blockIdx.x * GEMM_BM;
    const int n0   = blockIdx.y * GEMM_BN;

    v8f_t a00 = {}, a01 = {}, a10 = {}, a11 = {};
    const int ktiles = Kp / GEMM_KT;

    // Issue async copies for K-tile kt into LDS buffer `buf`.
    // 1536 b64 chunks total: 1024 for X (128 rows x 8), 512 for W (64 x 8).
    auto stage = [&](int kt, int buf) {
        const int kb = kt * GEMM_KT;
#pragma unroll
        for (int i = 0; i < 4; ++i) {          // X chunks
            int cid = tid + i * 256;
            int r = cid >> 3, c8 = cid & 7;
            unsigned lo = (unsigned)(uintptr_t)&lx[buf][r * GEMM_XS + c8 * 4];
            const void* ga = (const char*)Xb + (((size_t)(m0 + r) * Kp + kb) << 1) + c8 * 8;
            async_copy_b64(lo, ga);
        }
#pragma unroll
        for (int i = 0; i < 2; ++i) {          // W chunks
            int cid = tid + i * 256;
            int r = cid >> 3, c8 = cid & 7;
            unsigned lo = (unsigned)(uintptr_t)&lw[buf][r * GEMM_XS + c8 * 4];
            const void* ga = (const char*)Wb + (((size_t)(n0 + r) * Kp + kb) << 1) + c8 * 8;
            async_copy_b64(lo, ga);
        }
    };

    stage(0, 0);
    for (int kt = 0; kt < ktiles; ++kt) {
        const int buf = kt & 1;
        if (kt + 1 < ktiles) {
            stage(kt + 1, buf ^ 1);
            asm volatile("s_wait_asynccnt 0x6" ::: "memory");  // tile kt landed
        } else {
            asm volatile("s_wait_asynccnt 0x0" ::: "memory");
        }
        __syncthreads();   // all waves' tile-kt data visible

        union { unsigned u[8]; v16bf_t v; } A0, A1, B0, B1;
#pragma unroll
        for (int v = 0; v < 8; ++v) {
            int ka = 2 * v + 8 * g + (v >= 4 ? 8 : 0);       // A: K-pair index
            A0.u[v] = *(const unsigned*)&lx[buf][(wm + hl) * GEMM_XS + ka];
            A1.u[v] = *(const unsigned*)&lx[buf][(wm + 16 + hl) * GEMM_XS + ka];
            int kb2 = 2 * v + 8 * g + (v >= 4 ? 8 : 0);      // B: K-pair index
            B0.u[v] = *(const unsigned*)&lw[buf][(wn + hl) * GEMM_XS + kb2];
            B1.u[v] = *(const unsigned*)&lw[buf][(wn + 16 + hl) * GEMM_XS + kb2];
        }
        a00 = __builtin_amdgcn_wmma_f32_16x16x32_bf16(false, A0.v, false, B0.v, (short)0, a00, false, false);
        a01 = __builtin_amdgcn_wmma_f32_16x16x32_bf16(false, A0.v, false, B1.v, (short)0, a01, false, false);
        a10 = __builtin_amdgcn_wmma_f32_16x16x32_bf16(false, A1.v, false, B0.v, (short)0, a10, false, false);
        a11 = __builtin_amdgcn_wmma_f32_16x16x32_bf16(false, A1.v, false, B1.v, (short)0, a11, false, false);
        __syncthreads();   // done reading `buf` before it is refilled
    }

    // Epilogue: C/D layout -> row = r + 8*g (per 16-row tile), col = lane&15.
    const int col0 = n0 + wn + hl;
    const int col1 = col0 + 16;
    const float bv0 = bias[col0];
    const float bv1 = bias[col1];
#pragma unroll
    for (int r = 0; r < 8; ++r) {
        int row0 = m0 + wm + r + 8 * g;
        int row1 = row0 + 16;
        float y;
        y = a00[r] + bv0; Yb[(size_t)row0 * Nc + col0] = f2bf(y > 0.0f ? y : 0.0f);
        y = a01[r] + bv1; Yb[(size_t)row0 * Nc + col1] = f2bf(y > 0.0f ? y : 0.0f);
        y = a10[r] + bv0; Yb[(size_t)row1 * Nc + col0] = f2bf(y > 0.0f ? y : 0.0f);
        y = a11[r] + bv1; Yb[(size_t)row1 * Nc + col1] = f2bf(y > 0.0f ? y : 0.0f);
    }
}

// Max over K samples: in [groups*K, C] bf16 -> out [groups, C] bf16.
__global__ void maxpool_kernel(const unsigned short* __restrict__ Y, unsigned short* __restrict__ out,
                               int groups, int K, int C) {
    size_t total = (size_t)groups * C;
    size_t i = (size_t)blockIdx.x * blockDim.x + threadIdx.x;
    if (i >= total) return;
    size_t grp = i / C;
    int c = (int)(i % C);
    const unsigned short* p = Y + grp * (size_t)K * C + c;
    float m = bf2f(p[0]);
    for (int k = 1; k < K; ++k) m = fmaxf(m, bf2f(p[(size_t)k * C]));
    out[i] = f2bf(m);
}

// 3-NN inverse-distance interpolation + concat skip feats -> bf16 padded rows.
__global__ void knn3_interp_concat_kernel(const float* __restrict__ ux, const float* __restrict__ kx,
                                          const unsigned short* __restrict__ kf,
                                          const unsigned short* __restrict__ uf,
                                          unsigned short* __restrict__ Xb,
                                          int Nu, int Nk, int Ck, int Cu, int Kp) {
    int p = blockIdx.x * blockDim.x + threadIdx.x;
    if (p >= 2 * Nu) return;
    int b = p / Nu;
    const float* pu = ux + (size_t)p * 3;
    float cx = pu[0], cy = pu[1], cz = pu[2];
    const float* pk = kx + (size_t)b * Nk * 3;
    float d0 = 1e30f, d1 = 1e30f, d2 = 1e30f;
    int i0 = 0, i1 = 0, i2 = 0;
    for (int t = 0; t < Nk; ++t) {
        float dx = pk[t * 3 + 0] - cx;
        float dy = pk[t * 3 + 1] - cy;
        float dz = pk[t * 3 + 2] - cz;
        float d  = dx * dx + dy * dy + dz * dz;
        if (d < d0)      { d2 = d1; i2 = i1; d1 = d0; i1 = i0; d0 = d; i0 = t; }
        else if (d < d1) { d2 = d1; i2 = i1; d1 = d;  i1 = t; }
        else if (d < d2) { d2 = d;  i2 = t; }
    }
    float w0 = 1.0f / (d0 + 1e-8f);
    float w1 = 1.0f / (d1 + 1e-8f);
    float w2 = 1.0f / (d2 + 1e-8f);
    float ws = w0 + w1 + w2;
    w0 /= ws; w1 /= ws; w2 /= ws;
    const unsigned short* f0 = kf + ((size_t)b * Nk + i0) * Ck;
    const unsigned short* f1 = kf + ((size_t)b * Nk + i1) * Ck;
    const unsigned short* f2 = kf + ((size_t)b * Nk + i2) * Ck;
    unsigned short* xo = Xb + (size_t)p * Kp;
    for (int c = 0; c < Ck; ++c)
        xo[c] = f2bf(w0 * bf2f(f0[c]) + w1 * bf2f(f1[c]) + w2 * bf2f(f2[c]));
    const unsigned short* pf = uf + (size_t)p * Cu;
    for (int c = 0; c < Cu; ++c) xo[Ck + c] = pf[c];
    for (int c = Ck + Cu; c < Kp; ++c) xo[c] = 0;
}

// [B, Nu, C] bf16 -> [B, C, Nu] fp32 (final output).
__global__ void transpose_out_kernel(const unsigned short* __restrict__ X, float* __restrict__ out,
                                     int Nu, int C) {
    size_t total = (size_t)2 * Nu * C;
    size_t i = (size_t)blockIdx.x * blockDim.x + threadIdx.x;
    if (i >= total) return;
    size_t t = i;
    int u = (int)(t % Nu); t /= Nu;
    int c = (int)(t % C);
    int b = (int)(t / C);
    out[i] = bf2f(X[((size_t)b * Nu + u) * C + c]);
}

// ---------------------------------------------------------------------------
// Host orchestration
// ---------------------------------------------------------------------------
static inline unsigned blks(size_t n, int t) { return (unsigned)((n + t - 1) / t); }
static inline int rup32(int x) { return (x + 31) & ~31; }

extern "C" void kernel_launch(void* const* d_in, const int* in_sizes, int n_in,
                              void* d_out, int out_size, void* d_ws, size_t ws_size,
                              hipStream_t stream) {
    (void)in_sizes; (void)n_in; (void)out_size; (void)ws_size;

    // Flattening assumption: xyz, then per stage (sa1,sa2,sa3,fp3,fp2):
    // all weights, then all biases (params dict insertion order).
    const float* xyz = (const float*)d_in[0];
    const float* Wf[13] = {
        (const float*)d_in[1],  (const float*)d_in[2],  (const float*)d_in[3],   // sa1 w
        (const float*)d_in[7],  (const float*)d_in[8],  (const float*)d_in[9],   // sa2 w
        (const float*)d_in[13], (const float*)d_in[14], (const float*)d_in[15],  // sa3 w
        (const float*)d_in[19], (const float*)d_in[20],                          // fp3 w
        (const float*)d_in[23], (const float*)d_in[24]                           // fp2 w
    };
    const float* Bf[13] = {
        (const float*)d_in[4],  (const float*)d_in[5],  (const float*)d_in[6],
        (const float*)d_in[10], (const float*)d_in[11], (const float*)d_in[12],
        (const float*)d_in[16], (const float*)d_in[17], (const float*)d_in[18],
        (const float*)d_in[21], (const float*)d_in[22],
        (const float*)d_in[25], (const float*)d_in[26]
    };
    static const int LNc[13] = {64, 64, 128, 128, 128, 256, 256, 256, 512, 512, 256, 256, 128};
    static const int LKc[13] = {3,  64, 64,  131, 128, 128, 259, 256, 256, 768, 512, 384, 256};

    char* cur = (char*)d_ws;
    auto alloc = [&](size_t bytes) -> void* {
        void* r = (void*)cur;
        cur += (bytes + 255) & ~(size_t)255;
        return r;
    };
    int*   fidx1 = (int*)  alloc(sizeof(int)   * 2 * 4096);
    float* l1x   = (float*)alloc(sizeof(float) * 2 * 4096 * 3);
    int*   gidx1 = (int*)  alloc(sizeof(int)   * 2 * 4096 * 32);
    int*   fidx2 = (int*)  alloc(sizeof(int)   * 2 * 1024);
    float* l2x   = (float*)alloc(sizeof(float) * 2 * 1024 * 3);
    int*   gidx2 = (int*)  alloc(sizeof(int)   * 2 * 1024 * 64);
    int*   fidx3 = (int*)  alloc(sizeof(int)   * 2 * 512);
    float* l3x   = (float*)alloc(sizeof(float) * 2 * 512 * 3);
    int*   gidx3 = (int*)  alloc(sizeof(int)   * 2 * 512 * 128);
    unsigned short* l1f = (unsigned short*)alloc(sizeof(short) * 2 * 4096 * 128);
    unsigned short* l2f = (unsigned short*)alloc(sizeof(short) * 2 * 1024 * 256);
    unsigned short* l3f = (unsigned short*)alloc(sizeof(short) * 2 * 512 * 512);
    unsigned short* l2u = (unsigned short*)alloc(sizeof(short) * 2 * 1024 * 256);
    unsigned short* Wb[13];
    for (int l = 0; l < 13; ++l)
        Wb[l] = (unsigned short*)alloc(sizeof(short) * (size_t)LNc[l] * rup32(LKc[l]));
    unsigned short* P0 = (unsigned short*)alloc(sizeof(short) * (size_t)262144 * 288);
    unsigned short* P1 = (unsigned short*)alloc(sizeof(short) * (size_t)262144 * 512);

    // Pack all weights to padded bf16.
    for (int l = 0; l < 13; ++l) {
        size_t n = (size_t)LNc[l] * rup32(LKc[l]);
        pack_weight_kernel<<<blks(n, 256), 256, 0, stream>>>(Wf[l], Wb[l], LNc[l], LKc[l], rup32(LKc[l]));
    }

    auto gemm = [&](int l, const unsigned short* X, unsigned short* Y, int M) {
        dim3 grid((unsigned)(M / GEMM_BM), (unsigned)(LNc[l] / GEMM_BN));
        gemm_bias_relu_wmma<<<grid, 256, 0, stream>>>(X, Wb[l], Bf[l], Y, M, LNc[l], rup32(LKc[l]));
    };

    // ---- SA1: 16384 -> 4096 centers, K=32, MLP 3->64->64->128 ----
    fps_kernel<16384><<<2, 1024, 0, stream>>>(xyz, fidx1, 4096);
    gather_xyz_kernel<<<blks(2 * 4096, 256), 256, 0, stream>>>(xyz, fidx1, l1x, 16384, 4096);
    ball_query_kernel<<<blks(2 * 4096, 256), 256, 0, stream>>>(xyz, l1x, gidx1, 16384, 4096, 32, 0.04f);
    gather_group_kernel<<<blks((size_t)2 * 4096 * 32, 256), 256, 0, stream>>>(
        xyz, l1x, gidx1, l1f /*unused Cf=0*/, P0, 16384, 4096, 32, 0, 32);
    gemm(0, P0, P1, 262144);
    gemm(1, P1, P0, 262144);
    gemm(2, P0, P1, 262144);
    maxpool_kernel<<<blks((size_t)8192 * 128, 256), 256, 0, stream>>>(P1, l1f, 8192, 32, 128);

    // ---- SA2: 4096 -> 1024 centers, K=64, MLP 131->128->128->256 ----
    fps_kernel<4096><<<2, 1024, 0, stream>>>(l1x, fidx2, 1024);
    gather_xyz_kernel<<<blks(2 * 1024, 256), 256, 0, stream>>>(l1x, fidx2, l2x, 4096, 1024);
    ball_query_kernel<<<blks(2 * 1024, 256), 256, 0, stream>>>(l1x, l2x, gidx2, 4096, 1024, 64, 0.16f);
    gather_group_kernel<<<blks((size_t)2 * 1024 * 64, 256), 256, 0, stream>>>(
        l1x, l2x, gidx2, l1f, P0, 4096, 1024, 64, 128, 160);
    gemm(3, P0, P1, 131072);
    gemm(4, P1, P0, 131072);
    gemm(5, P0, P1, 131072);
    maxpool_kernel<<<blks((size_t)2048 * 256, 256), 256, 0, stream>>>(P1, l2f, 2048, 64, 256);

    // ---- SA3: 1024 -> 512 centers, K=128, MLP 259->256->256->512 ----
    fps_kernel<1024><<<2, 1024, 0, stream>>>(l2x, fidx3, 512);
    gather_xyz_kernel<<<blks(2 * 512, 256), 256, 0, stream>>>(l2x, fidx3, l3x, 1024, 512);
    ball_query_kernel<<<blks(2 * 512, 256), 256, 0, stream>>>(l2x, l3x, gidx3, 1024, 512, 128, 0.36f);
    gather_group_kernel<<<blks((size_t)2 * 512 * 128, 256), 256, 0, stream>>>(
        l2x, l3x, gidx3, l2f, P0, 1024, 512, 128, 256, 288);
    gemm(6, P0, P1, 131072);
    gemm(7, P1, P0, 131072);
    gemm(8, P0, P1, 131072);
    maxpool_kernel<<<blks((size_t)1024 * 512, 256), 256, 0, stream>>>(P1, l3f, 1024, 128, 512);

    // ---- FP3: interpolate l3 -> l2 (Nu=1024, Nk=512), MLP 768->512->256 ----
    knn3_interp_concat_kernel<<<blks(2 * 1024, 256), 256, 0, stream>>>(
        l2x, l3x, l3f, l2f, P0, 1024, 512, 512, 256, 768);
    gemm(9, P0, P1, 2048);
    gemm(10, P1, l2u, 2048);

    // ---- FP2: interpolate l2u -> l1 (Nu=4096, Nk=1024), MLP 384->256->128 ----
    knn3_interp_concat_kernel<<<blks(2 * 4096, 256), 256, 0, stream>>>(
        l1x, l2x, l2u, l1f, P0, 4096, 1024, 256, 128, 384);
    gemm(11, P0, P1, 8192);
    gemm(12, P1, P0, 8192);

    // ---- Output: [B,4096,128] -> [B,128,4096] fp32 ----
    transpose_out_kernel<<<blks((size_t)2 * 128 * 4096, 256), 256, 0, stream>>>(
        P0, (float*)d_out, 4096, 128);
}
